// RealMambaNet_29729763623373
// MI455X (gfx1250) — matrix operand order repaired and checked by reading, not compiled, not run on hardware
//
#include <hip/hip_runtime.h>
#include <hip/hip_bf16.h>
#include <math.h>

// RealMambaNet forward for MI455X (gfx1250), wave32 + WMMA f16->f32.
// GEMMs: v_wmma_f32_16x16x32_f16, 4-wave workgroups, B panel staged into
// double-buffered LDS by the Tensor Data Mover (TENSOR_LOAD_TO_LDS +
// s_wait_tensorcnt), A operand streamed with global_load_b128.

#define B_   8
#define T_   512
#define F_   128
#define DM_  384
#define DS_  16
#define DC_  4
#define NL_  12
#define DI_  768
#define DTR_ 24
#define L_   256
#define NC_  4
#define ROWS (B_ * L_)   // 2048 token rows

typedef __attribute__((ext_vector_type(16))) _Float16 v16h;
typedef __attribute__((ext_vector_type(8)))  _Float16 v8h;
typedef __attribute__((ext_vector_type(8)))  float    v8f;

#if defined(__has_builtin)
#if __has_builtin(__builtin_amdgcn_tensor_load_to_lds) && \
    __has_builtin(__builtin_amdgcn_s_wait_tensorcnt)
#define USE_TDM 1
#endif
#endif
#ifndef USE_TDM
#define USE_TDM 0
#endif

#if USE_TDM
typedef unsigned tdm_v4u __attribute__((ext_vector_type(4)));
typedef int      tdm_v8i __attribute__((ext_vector_type(8)));
typedef int      tdm_v4i __attribute__((ext_vector_type(4)));

// Issue a TDM copy of a 64-row x 32-half (4KB) weight chunk into LDS.
// D# per CDNA5 ISA 8.3/8.4: 2D tile, data_size=2B, tile_dim0=32 (contig K),
// tile_dim1=64 (rows), tensor_dim0_stride=K halfs, descriptor type=2.
__device__ __forceinline__ void tdm_load_b_panel(const _Float16* gsrc,
                                                 unsigned lds_addr, int K) {
  unsigned long long ga = (unsigned long long)(size_t)gsrc;
  tdm_v4u g0;
  g0[0] = 1u;                                      // count=1 valid descriptor
  g0[1] = lds_addr;                                // LDS byte address
  g0[2] = (unsigned)(ga & 0xFFFFFFFFu);            // global_addr[31:0]
  g0[3] = (unsigned)((ga >> 32) & 0x01FFFFFFu)     // global_addr[56:32]
          | (2u << 30);                            // type=2 ("image")
  unsigned td0 = (unsigned)K;                      // tensor_dim0 (halfs)
  unsigned td1 = 1u << 16;                         // tensor_dim1 (big)
  tdm_v8i g1;
  g1[0] = 0x10000;                                 // data_size=1 -> 2 bytes
  g1[1] = (int)((td0 & 0xFFFFu) << 16);            // tensor_dim0 lo
  g1[2] = (int)((td0 >> 16) | ((td1 & 0xFFFFu) << 16));
  g1[3] = (int)((td1 >> 16) | (32u << 16));        // tile_dim0 = 32
  g1[4] = 64;                                      // tile_dim1=64, tile_dim2=0
  g1[5] = K;                                       // tensor_dim0_stride lo32
  g1[6] = 0;
  g1[7] = 0;
  tdm_v4i z4 = {0, 0, 0, 0};
  tdm_v8i z8 = {0, 0, 0, 0, 0, 0, 0, 0};
  __builtin_amdgcn_tensor_load_to_lds(g0, g1, z4, z4, z8, 0);
}
#endif

// ---------------------------------------------------------------------------
// f32 -> f16 weight convert with zero padding: src [layers, N, K] row-major
// -> dst [layers, Npad, Kpad], zeros outside [N,K].
// ---------------------------------------------------------------------------
__global__ __launch_bounds__(256) void cvt_pad_k(
    const float* __restrict__ src, _Float16* __restrict__ dst,
    int layers, int N, int K, int Npad, int Kpad) {
  size_t total = (size_t)layers * Npad * Kpad;
  size_t idx = (size_t)blockIdx.x * 256 + threadIdx.x;
  if (idx >= total) return;
  int kp = (int)(idx % Kpad);
  size_t t = idx / Kpad;
  int np = (int)(t % Npad);
  int l  = (int)(t / Npad);
  float v = (np < N && kp < K) ? src[((size_t)l * N + np) * K + kp] : 0.f;
  dst[idx] = (_Float16)v;
}

// ---------------------------------------------------------------------------
// Patch embed: tok[b,l,d] = sum_{p,q} x[b, hb*16+p, wb*16+q] * pw[d,p,q]
//              + pb[d] + pos[l,d], with l = hb*8 + wb.  (f32 master)
// ---------------------------------------------------------------------------
__global__ __launch_bounds__(256) void patch_embed_k(
    const float* __restrict__ x, const float* __restrict__ pw,
    const float* __restrict__ pb, const float* __restrict__ pos,
    float* __restrict__ tok) {
  int idx = blockIdx.x * 256 + threadIdx.x;        // (b*L + l)*DM + d
  if (idx >= ROWS * DM_) return;
  int d  = idx % DM_;
  int bl = idx / DM_;
  int l  = bl % L_;
  int b  = bl / L_;
  int hb = l / 8, wb = l % 8;                       // 32 x 8 patches
  const float* xb = x + ((size_t)b * T_ + hb * 16) * F_ + wb * 16;
  const float* w  = pw + (size_t)d * 256;
  float acc = 0.f;
  #pragma unroll
  for (int p = 0; p < 16; ++p)
    #pragma unroll 4
    for (int q = 0; q < 16; ++q)
      acc += xb[p * F_ + q] * w[p * 16 + q];
  tok[idx] = acc + pb[d] + pos[(size_t)l * DM_ + d];
}

// ---------------------------------------------------------------------------
// LayerNorm over last dim (384). One wave32 per row, 12 elems per lane.
// ---------------------------------------------------------------------------
template <typename OutT>
__global__ __launch_bounds__(32) void layernorm_k(
    const float* __restrict__ in, const float* __restrict__ g,
    const float* __restrict__ bta, OutT* __restrict__ out) {
  int row = blockIdx.x;
  const float* p = in + (size_t)row * DM_;
  float v[12];
  float s = 0.f;
  #pragma unroll
  for (int i = 0; i < 12; ++i) { v[i] = p[threadIdx.x + 32 * i]; s += v[i]; }
  #pragma unroll
  for (int off = 16; off; off >>= 1) s += __shfl_xor(s, off, 32);
  float mean = s * (1.f / DM_);
  float ss = 0.f;
  #pragma unroll
  for (int i = 0; i < 12; ++i) { float d = v[i] - mean; ss += d * d; }
  #pragma unroll
  for (int off = 16; off; off >>= 1) ss += __shfl_xor(ss, off, 32);
  float rstd = rsqrtf(ss * (1.f / DM_) + 1e-5f);
  OutT* o = out + (size_t)row * DM_;
  #pragma unroll
  for (int i = 0; i < 12; ++i) {
    int c = threadIdx.x + 32 * i;
    o[c] = (OutT)((v[i] - mean) * rstd * g[c] + bta[c]);
  }
}

__device__ __forceinline__ float softplus_f(float x) {
  return x > 20.f ? x : log1pf(__expf(x));
}

// ---------------------------------------------------------------------------
// WMMA GEMM: C[M,N] (+)= act( A[M,K] @ Bw[N,K]^T + bias ), f16 in, f32 acc.
// 4-wave workgroup computes a 128x64 tile; wave w owns rows m0+w*32..+31
// (2 M-frags x 4 N-frags = 8 v_wmma per K-step per wave). The shared 64x32
// B chunk is staged into double-buffered LDS by the TDM (wave 0 issues
// TENSOR_LOAD_TO_LDS for chunk s+1 while everyone computes chunk s), and
// read back as ds_load_b128 fragments.
// Requires: M % 128 == 0, N % 64 == 0, K % 32 == 0 (weights pre-padded).
// Fragment layouts per CDNA5 ISA 7.12.2 (wave32).
// ---------------------------------------------------------------------------
template <typename OutT, int ACT, int ADDRES>
__global__ __launch_bounds__(128) void gemm_wmma_k(
    const _Float16* __restrict__ A, int lda,
    const _Float16* __restrict__ Bw,           // [N, K] f16 (padded)
    const float* __restrict__ bias,            // length N, or nullptr
    OutT* __restrict__ C, int ldc, int K) {
  __shared__ _Float16 Bs[2][64 * 32];          // 2 x 4KB double buffer

  int tid  = threadIdx.x;
  int wave = tid >> 5;
  int lane = tid & 31;
  int r    = lane & 15;
  int hh   = lane >> 4;
  int m0   = blockIdx.x * 128 + wave * 32;
  int n0   = blockIdx.y * 64;

  const _Float16* arow0  = A + (size_t)(m0 + r) * lda + hh * 8;
  const _Float16* arow1  = arow0 + (size_t)16 * lda;
  const _Float16* bpanel = Bw + (size_t)n0 * K;

  int nsteps = K / 32;

#if USE_TDM
  unsigned lds0 = (unsigned)(size_t)&Bs[0][0];
  unsigned lds1 = (unsigned)(size_t)&Bs[1][0];
  if (wave == 0) tdm_load_b_panel(bpanel, lds0, K);
#endif

  v8f acc[2][4] = {};
  for (int s = 0; s < nsteps; ++s) {
    int kb  = s * 32;
    int buf = s & 1;
#if USE_TDM
    if (wave == 0) {
      if (s + 1 < nsteps) {
        tdm_load_b_panel(bpanel + (size_t)(kb + 32), buf ? lds0 : lds1, K);
        __builtin_amdgcn_s_wait_tensorcnt(1);   // current chunk done (in-order)
      } else {
        __builtin_amdgcn_s_wait_tensorcnt(0);   // last chunk done
      }
    }
#else
    // Fallback: cooperative synchronous copy of the 4KB chunk (2 x b128/thread).
    {
      int buf0 = buf;
      #pragma unroll
      for (int j = 0; j < 2; ++j) {
        int idx = tid + j * 128;                      // 0..255, 8 halfs each
        const v8h* src = (const v8h*)(bpanel + (size_t)(idx >> 2) * K + kb +
                                      (idx & 3) * 8);
        *(v8h*)&Bs[buf0][idx * 8] = *src;
      }
    }
#endif
    __syncthreads();

    v8h a0lo = *(const v8h*)(arow0 + kb);
    v8h a0hi = *(const v8h*)(arow0 + kb + 16);
    v8h a1lo = *(const v8h*)(arow1 + kb);
    v8h a1hi = *(const v8h*)(arow1 + kb + 16);
    v16h af0 = __builtin_shufflevector(a0lo, a0hi,
        0,1,2,3,4,5,6,7,8,9,10,11,12,13,14,15);
    v16h af1 = __builtin_shufflevector(a1lo, a1hi,
        0,1,2,3,4,5,6,7,8,9,10,11,12,13,14,15);
    #pragma unroll
    for (int t = 0; t < 4; ++t) {
      const v8h* bls = (const v8h*)&Bs[buf][(t * 16 + r) * 32 + hh * 8];
      v8h b0 = bls[0];
      v8h b1 = bls[2];
      v16h bf = __builtin_shufflevector(b0, b1,
          0,1,2,3,4,5,6,7,8,9,10,11,12,13,14,15);
      acc[0][t] = __builtin_amdgcn_wmma_f32_16x16x32_f16(
          false, af0, false, bf, (short)0, acc[0][t], false, false);
      acc[1][t] = __builtin_amdgcn_wmma_f32_16x16x32_f16(
          false, af1, false, bf, (short)0, acc[1][t], false, false);
    }
    __syncthreads();
  }

  #pragma unroll
  for (int mm = 0; mm < 2; ++mm) {
    #pragma unroll
    for (int t = 0; t < 4; ++t) {
      int n = n0 + t * 16 + r;
      float bv = bias ? bias[n] : 0.f;
      #pragma unroll
      for (int rr = 0; rr < 8; ++rr) {
        int mr = m0 + mm * 16 + hh * 8 + rr;
        float vv = acc[mm][t][rr] + bv;
        if (ACT == 1) vv = softplus_f(vv);
        size_t off = (size_t)mr * ldc + n;
        if (ADDRES) vv += (float)C[off];
        C[off] = (OutT)vv;
      }
    }
  }
}

// ---------------------------------------------------------------------------
// Causal depthwise conv1d (kernel 4) over L within each batch, + bias + SiLU.
// u = xz[:, 0:DI] (f16, row stride 2*DI). Output f16.
// ---------------------------------------------------------------------------
__global__ __launch_bounds__(256) void conv_silu_k(
    const _Float16* __restrict__ xz, const float* __restrict__ cw,
    const float* __restrict__ cb, _Float16* __restrict__ out) {
  int idx = blockIdx.x * 256 + threadIdx.x;
  if (idx >= ROWS * DI_) return;
  int e    = idx % DI_;
  int row  = idx / DI_;
  int l    = row % L_;
  int base = row - l;                 // b*L
  float acc = cb[e];
  #pragma unroll
  for (int k = 0; k < DC_; ++k) {
    int ls = l - (DC_ - 1) + k;
    if (ls >= 0)
      acc += (float)xz[(size_t)(base + ls) * (2 * DI_) + e] * cw[e * DC_ + k];
  }
  float sg = 1.f / (1.f + __expf(-acc));
  out[idx] = (_Float16)(acc * sg);
}

// ---------------------------------------------------------------------------
// Selective-state scan. One thread per (b, channel e): 16-wide state in
// registers; B_t / C_t broadcast through LDS per timestep.
// Fuses D-skip and the silu(z) gate. Writes y (f16) for the out_proj GEMM.
// ---------------------------------------------------------------------------
__global__ __launch_bounds__(256) void scan_k(
    const _Float16* __restrict__ delta, const _Float16* __restrict__ u,
    const _Float16* __restrict__ xz,    const _Float16* __restrict__ projp,
    const float* __restrict__ A_log,    const float* __restrict__ Dsk,
    _Float16* __restrict__ y) {
  int b = blockIdx.y;
  int e = blockIdx.x * 256 + threadIdx.x;       // < DI_
  __shared__ float Bs[DS_], Cs[DS_];
  float Arow[DS_], hst[DS_];
  #pragma unroll
  for (int n = 0; n < DS_; ++n) {
    Arow[n] = -__expf(A_log[(size_t)e * DS_ + n]);
    hst[n]  = 0.f;
  }
  float dsk = Dsk[e];
  for (int l = 0; l < L_; ++l) {
    int row = b * L_ + l;
    if (threadIdx.x < DS_)
      Bs[threadIdx.x] = (float)projp[(size_t)row * 64 + DTR_ + threadIdx.x];
    else if (threadIdx.x < 2 * DS_)
      Cs[threadIdx.x - DS_] =
          (float)projp[(size_t)row * 64 + DTR_ + DS_ + (threadIdx.x - DS_)];
    __syncthreads();
    float d  = (float)delta[(size_t)row * DI_ + e];
    float ut = (float)u[(size_t)row * DI_ + e];
    float zt = (float)xz[(size_t)row * (2 * DI_) + DI_ + e];
    float du = d * ut;
    float accv = 0.f;
    #pragma unroll
    for (int n = 0; n < DS_; ++n) {
      float dA = __expf(d * Arow[n]);
      hst[n] = dA * hst[n] + du * Bs[n];
      accv  += hst[n] * Cs[n];
    }
    float yv = accv + ut * dsk;
    float sg = 1.f / (1.f + __expf(-zt));
    y[(size_t)row * DI_ + e] = (_Float16)(yv * (zt * sg));
    __syncthreads();
  }
}

// ---------------------------------------------------------------------------
// Mean-pool over L then 4-way head. One block per batch.
// ---------------------------------------------------------------------------
__global__ __launch_bounds__(128) void pool_head_k(
    const float* __restrict__ lnf, const float* __restrict__ hw,
    const float* __restrict__ hb, float* __restrict__ out) {
  int b = blockIdx.x;
  __shared__ float pooled[DM_];
  for (int d = threadIdx.x; d < DM_; d += 128) {
    float s = 0.f;
    for (int l = 0; l < L_; ++l)
      s += lnf[((size_t)b * L_ + l) * DM_ + d];
    pooled[d] = s * (1.f / L_);
  }
  __syncthreads();
  if (threadIdx.x < NC_) {
    float s = hb[threadIdx.x];
    for (int d = 0; d < DM_; ++d)
      s += pooled[d] * hw[threadIdx.x * DM_ + d];
    out[b * NC_ + threadIdx.x] = s;
  }
}

// ---------------------------------------------------------------------------
extern "C" void kernel_launch(void* const* d_in, const int* in_sizes, int n_in,
                              void* d_out, int out_size, void* d_ws, size_t ws_size,
                              hipStream_t stream) {
  const float* x      = (const float*)d_in[0];
  const float* pw     = (const float*)d_in[1];
  const float* pb     = (const float*)d_in[2];
  const float* pos    = (const float*)d_in[3];
  const float* ln_g   = (const float*)d_in[4];
  const float* ln_b   = (const float*)d_in[5];
  const float* Wi_all = (const float*)d_in[6];
  const float* cw_all = (const float*)d_in[7];
  const float* cb_all = (const float*)d_in[8];
  const float* Wx_all = (const float*)d_in[9];
  const float* Wdt_all= (const float*)d_in[10];
  const float* bdt_all= (const float*)d_in[11];
  const float* Al_all = (const float*)d_in[12];
  const float* Dsk_all= (const float*)d_in[13];
  const float* Wo_all = (const float*)d_in[14];
  const float* fn_g   = (const float*)d_in[15];
  const float* fn_b   = (const float*)d_in[16];
  const float* head_w = (const float*)d_in[17];
  const float* head_b = (const float*)d_in[18];

  // ---- workspace layout ----
  float* tok = (float*)d_ws;                         // 2048*384 f32
  float* lnf = tok + (size_t)ROWS * DM_;             // 2048*384 f32
  _Float16* hln_h   = (_Float16*)(lnf + (size_t)ROWS * DM_);
  _Float16* xz_h    = hln_h   + (size_t)ROWS * DM_;          // 2048*1536
  _Float16* uconv_h = xz_h    + (size_t)ROWS * 2 * DI_;      // 2048*768
  _Float16* projp_h = uconv_h + (size_t)ROWS * DI_;          // 2048*64
  _Float16* delta_h = projp_h + (size_t)ROWS * 64;           // 2048*768
  _Float16* ybuf_h  = delta_h + (size_t)ROWS * DI_;          // 2048*768
  _Float16* wi_h    = ybuf_h  + (size_t)ROWS * DI_;          // 12*1536*384
  _Float16* wx_h    = wi_h    + (size_t)NL_ * 2 * DI_ * DM_; // 12*64*768
  _Float16* wdt_h   = wx_h    + (size_t)NL_ * 64 * DI_;      // 12*768*32
  _Float16* wo_h    = wdt_h   + (size_t)NL_ * DI_ * 32;      // 12*384*768
  _Float16* ws_end  = wo_h    + (size_t)NL_ * DM_ * DI_;
  size_t need = (size_t)((char*)ws_end - (char*)d_ws);
  if (ws_size < need) return;

  // ---- weight conversion (f32 -> f16, zero-padded) ----
  {
    size_t n;
    n = (size_t)NL_ * 2 * DI_ * DM_;
    cvt_pad_k<<<(unsigned)((n + 255) / 256), 256, 0, stream>>>(
        Wi_all, wi_h, NL_, 2 * DI_, DM_, 2 * DI_, DM_);
    n = (size_t)NL_ * 64 * DI_;
    cvt_pad_k<<<(unsigned)((n + 255) / 256), 256, 0, stream>>>(
        Wx_all, wx_h, NL_, DTR_ + 2 * DS_, DI_, 64, DI_);
    n = (size_t)NL_ * DI_ * 32;
    cvt_pad_k<<<(unsigned)((n + 255) / 256), 256, 0, stream>>>(
        Wdt_all, wdt_h, NL_, DI_, DTR_, DI_, 32);
    n = (size_t)NL_ * DM_ * DI_;
    cvt_pad_k<<<(unsigned)((n + 255) / 256), 256, 0, stream>>>(
        Wo_all, wo_h, NL_, DM_, DI_, DM_, DI_);
  }

  patch_embed_k<<<(ROWS * DM_ + 255) / 256, 256, 0, stream>>>(x, pw, pb, pos, tok);

  for (int i = 0; i < NL_; ++i) {
    const float*    lng = ln_g   + (size_t)i * DM_;
    const float*    lnb = ln_b   + (size_t)i * DM_;
    const _Float16* Wi  = wi_h   + (size_t)i * 2 * DI_ * DM_;
    const float*    cwp = cw_all + (size_t)i * DI_ * DC_;
    const float*    cbp = cb_all + (size_t)i * DI_;
    const _Float16* Wx  = wx_h   + (size_t)i * 64 * DI_;
    const _Float16* Wdt = wdt_h  + (size_t)i * DI_ * 32;
    const float*    bdt = bdt_all+ (size_t)i * DI_;
    const float*    Alg = Al_all + (size_t)i * DI_ * DS_;
    const float*    Dsk = Dsk_all+ (size_t)i * DI_;
    const _Float16* Wo  = wo_h   + (size_t)i * DM_ * DI_;

    // pre-LN (f16 out for GEMM A)
    layernorm_k<_Float16><<<ROWS, 32, 0, stream>>>(tok, lng, lnb, hln_h);
    // in_proj: xz[2048,1536] = hln @ Wi^T
    gemm_wmma_k<_Float16, 0, 0><<<dim3(ROWS / 128, (2 * DI_) / 64), 128, 0, stream>>>(
        hln_h, DM_, Wi, nullptr, xz_h, 2 * DI_, DM_);
    // depthwise causal conv + SiLU on u = xz[:, :DI]
    conv_silu_k<<<(ROWS * DI_ + 255) / 256, 256, 0, stream>>>(
        xz_h, cwp, cbp, uconv_h);
    // x_proj: projp[2048,64pad] = uconv @ Wx^T (rows 56..63 of Wx are zero)
    gemm_wmma_k<_Float16, 0, 0><<<dim3(ROWS / 128, 1), 128, 0, stream>>>(
        uconv_h, DI_, Wx, nullptr, projp_h, 64, DI_);
    // dt_proj + softplus: delta = softplus(projp[:, :32pad] @ Wdt^T + bdt)
    // (Wdt cols 24..31 are zero, so Bm garbage in projp cols 24..31 is inert)
    gemm_wmma_k<_Float16, 1, 0><<<dim3(ROWS / 128, DI_ / 64), 128, 0, stream>>>(
        projp_h, 64, Wdt, bdt, delta_h, DI_, 32);
    // selective scan + D-skip + silu(z) gate
    scan_k<<<dim3(DI_ / 256, B_), 256, 0, stream>>>(
        delta_h, uconv_h, xz_h, projp_h, Alg, Dsk, ybuf_h);
    // out_proj with fused residual: tok += ybuf @ Wo^T  (f32 accumulate)
    gemm_wmma_k<float, 0, 1><<<dim3(ROWS / 128, DM_ / 64), 128, 0, stream>>>(
        ybuf_h, DI_, Wo, nullptr, tok, DM_, DI_);
  }

  layernorm_k<float><<<ROWS, 32, 0, stream>>>(tok, fn_g, fn_b, lnf);
  pool_head_k<<<B_, 128, 0, stream>>>(lnf, head_w, head_b, (float*)d_out);
}
